// Encoder_25280177504694
// MI455X (gfx1250) — compile-verified, matrix operand order checked by reading, not run on hardware
//
#include <hip/hip_runtime.h>
#include <hip/hip_bf16.h>

// ---------------------------------------------------------------------------
// Bidirectional 2-layer GRU encoder for MI455X (gfx1250).
//  - prep kernel: fp32 -> bf16 weight conversion + WMMA B-fragment packing
//  - main kernel: 2 persistent workgroups (one per direction), 32 waves each,
//    v_wmma_f32_16x16x32_bf16 for all recurrent GEMMs, hidden state double-
//    buffered in LDS (4 x 64KB), fused GRU gate epilogue per wave.
// ---------------------------------------------------------------------------

typedef __attribute__((ext_vector_type(16))) __bf16 v16bf;
typedef __attribute__((ext_vector_type(8)))  float  v8f;

#define BB 32        // batch
#define SS 512       // sequence
#define UU 1024      // units
#define G3 3072      // 3*U
#define KTILES 32    // 1024 / 32 (K per WMMA)
#define NTILES 192   // 3072 / 16
#define HSZ (BB * UU)   // one hidden-state buffer: 32768 bf16 = 64 KB

__device__ __forceinline__ float bf2f(unsigned short u) {
  union { unsigned u; float f; } x; x.u = ((unsigned)u) << 16; return x.f;
}
__device__ __forceinline__ unsigned short f2bf(float f) {
  union { float f; unsigned u; } x; x.f = f;
  unsigned u = (x.u + 0x7FFFu + ((x.u >> 16) & 1u)) >> 16;
  return (unsigned short)u;
}
__device__ __forceinline__ float sigm(float x) {
  return 1.0f / (1.0f + __expf(-x));
}

// ---------------------------------------------------------------------------
// Pack one (1024 x 3072) fp32 row-major matrix into bf16 WMMA-B fragment
// order: [ktile][ntile][lane][16 halves], each lane's 16 values contiguous
// (32B) so the main kernel loads a B fragment with two global_load_b128.
// K-within-lane follows the 16-bit operand layout:
//   lanes 0-15: K = {0..7, 16..23}; lanes 16-31: K = {8..15, 24..31}
// ---------------------------------------------------------------------------
__global__ void pack_weights(const float* __restrict__ W,
                             unsigned short* __restrict__ P) {
  const int tile = blockIdx.x;             // kk*NTILES + nt
  const int kk   = tile / NTILES;
  const int nt   = tile % NTILES;
  const int lane = threadIdx.x & 31;
  const int group = lane >> 4;
  const int col  = nt * 16 + (lane & 15);
  unsigned short* out = P + ((size_t)tile * 32 + lane) * 16;
#pragma unroll
  for (int v = 0; v < 8; ++v) {
    const int kb = (v < 4 ? 2 * v : 16 + 2 * (v - 4)) + 8 * group;
#pragma unroll
    for (int p = 0; p < 2; ++p) {
      const int k = kk * 32 + kb + p;
      out[v * 2 + p] = f2bf(W[(size_t)k * G3 + col]);
    }
  }
}

// A fragment (16x32 bf16) from an LDS hidden-state buffer [32 rows][1024 cols].
// Lane holds row mt*16 + (lane&15); lane group selects the K subset. The two
// 8-half groups are each 16B contiguous -> two ds_load_b128.
__device__ __forceinline__ v16bf load_a(const unsigned short* h,
                                        int row, int kk, int group) {
  union { v16bf v; uint4 q[2]; } r;
  const unsigned short* base = h + row * UU + kk * 32 + group * 8;
  r.q[0] = *(const uint4*)(base);
  r.q[1] = *(const uint4*)(base + 16);
  return r.v;
}

// B fragment (32x16 bf16) from the packed weight image: 32B contiguous/lane.
__device__ __forceinline__ v16bf load_b(const unsigned short* __restrict__ P,
                                        int kk, int nt, int lane) {
  union { v16bf v; uint4 q[2]; } r;
  const uint4* p = (const uint4*)(P + ((size_t)(kk * NTILES + nt) * 32 + lane) * 16);
  r.q[0] = p[0];
  r.q[1] = p[1];
  return r.v;
}

#define WMMA_BF16(A, Bf, C) \
  __builtin_amdgcn_wmma_f32_16x16x32_bf16(false, (A), false, (Bf), (short)0, (C), false, false)

struct Args {
  const int* tokens;
  const float* W1[2];
  const float* b1[2];   // (2,3072): [0]=input bias, [1]=recurrent bias
  const float* b2[2];
  const unsigned short* PR1[2];
  const unsigned short* PW2[2];
  const unsigned short* PR2[2];
  float* out;           // (B, S, 2048)
  float* state[2];      // state1 (fwd), state2 (bwd): each (B, 2048)
};

__global__ __launch_bounds__(1024, 1) void gru_encoder(Args a) {
  extern __shared__ unsigned short lds[];
  // LDS layout (offsets in bf16 elements, computed at runtime — no pointer
  // aggregates initialized from LDS, which the linker can't relocate):
  //   [0*HSZ) h1 buf0   [1*HSZ) h1 buf1   [2*HSZ) h2 buf0   [3*HSZ) h2 buf1

  const int dir = blockIdx.x;   // 0 = forward, 1 = backward
  const int* __restrict__ tok = a.tokens;
  const float* __restrict__ W1 = a.W1[dir];
  const float* __restrict__ b1 = a.b1[dir];
  const float* __restrict__ b2 = a.b2[dir];
  const unsigned short* __restrict__ PR1 = a.PR1[dir];
  const unsigned short* __restrict__ PW2 = a.PW2[dir];
  const unsigned short* __restrict__ PR2 = a.PR2[dir];

  const int tid   = threadIdx.x;
  const int lane  = tid & 31;
  const int w     = tid >> 5;      // wave id 0..31
  const int group = lane >> 4;
  const int nl    = lane & 15;

  for (int i = tid; i < HSZ; i += 1024) { lds[i] = 0; lds[2 * HSZ + i] = 0; }
  __syncthreads();

  int cur = 0;
  for (int t = 0; t < SS; ++t) {
    const int tt = dir ? (SS - 1 - t) : t;   // token index & output position
    unsigned short* h1c = lds + cur * HSZ;
    unsigned short* h1n = lds + (cur ^ 1) * HSZ;
    unsigned short* h2c = lds + (2 + cur) * HSZ;
    unsigned short* h2n = lds + (2 + (cur ^ 1)) * HSZ;

    // ---------------- layer 1: g1 = h1 @ R1 ; fused GRU with embedding x1 ---
#pragma unroll 1
    for (int i = 0; i < 4; ++i) {
      const int p  = w * 4 + i;     // 0..127 -> (mtile, jtile)
      const int mt = p >> 6;
      const int j  = p & 63;
      const int arow = mt * 16 + nl;
      v8f az = {}, ar = {}, ah = {};
#pragma unroll 4
      for (int kk = 0; kk < KTILES; ++kk) {
        const v16bf afr = load_a(h1c, arow, kk, group);
        az = WMMA_BF16(afr, load_b(PR1, kk, j,        lane), az);
        ar = WMMA_BF16(afr, load_b(PR1, kk, 64 + j,   lane), ar);
        ah = WMMA_BF16(afr, load_b(PR1, kk, 128 + j,  lane), ah);
      }
      const int col = j * 16 + nl;
#pragma unroll
      for (int v = 0; v < 8; ++v) {
        const int row = mt * 16 + v + 8 * group;            // C-layout row
        const int tkn = tok[row * SS + tt];
        const float zi = W1[(size_t)tkn * G3 + col]        + b1[col];
        const float ri = W1[(size_t)tkn * G3 + 1024 + col] + b1[1024 + col];
        const float hi = W1[(size_t)tkn * G3 + 2048 + col] + b1[2048 + col];
        const float z  = sigm(zi + az[v] + b1[G3 + col]);
        const float r  = sigm(ri + ar[v] + b1[G3 + 1024 + col]);
        const float hh = tanhf(hi + r * (ah[v] + b1[G3 + 2048 + col]));
        const float ho = bf2f(h1c[row * UU + col]);
        h1n[row * UU + col] = f2bf(z * ho + (1.0f - z) * hh);
      }
    }
    __syncthreads();

    // ------------- layer 2: x2 = h1_new @ W2 + b2[0]; g2 = h2 @ R2 + b2[1] --
#pragma unroll 1
    for (int i = 0; i < 4; ++i) {
      const int p  = w * 4 + i;
      const int mt = p >> 6;
      const int j  = p & 63;
      const int arow = mt * 16 + nl;
      v8f xz = {}, xr = {}, xh = {}, gz = {}, gr = {}, gh = {};
#pragma unroll 2
      for (int kk = 0; kk < KTILES; ++kk) {
        const v16bf a1 = load_a(h1n, arow, kk, group);
        const v16bf a2 = load_a(h2c, arow, kk, group);
        xz = WMMA_BF16(a1, load_b(PW2, kk, j,       lane), xz);
        xr = WMMA_BF16(a1, load_b(PW2, kk, 64 + j,  lane), xr);
        xh = WMMA_BF16(a1, load_b(PW2, kk, 128 + j, lane), xh);
        gz = WMMA_BF16(a2, load_b(PR2, kk, j,       lane), gz);
        gr = WMMA_BF16(a2, load_b(PR2, kk, 64 + j,  lane), gr);
        gh = WMMA_BF16(a2, load_b(PR2, kk, 128 + j, lane), gh);
      }
      const int col = j * 16 + nl;
#pragma unroll
      for (int v = 0; v < 8; ++v) {
        const int row = mt * 16 + v + 8 * group;
        const float z  = sigm(xz[v] + b2[col]        + gz[v] + b2[G3 + col]);
        const float r  = sigm(xr[v] + b2[1024 + col] + gr[v] + b2[G3 + 1024 + col]);
        const float hh = tanhf(xh[v] + b2[2048 + col] +
                               r * (gh[v] + b2[G3 + 2048 + col]));
        const float ho = bf2f(h2c[row * UU + col]);
        const float hn = z * ho + (1.0f - z) * hh;
        h2n[row * UU + col] = f2bf(hn);
        a.out[((size_t)row * SS + tt) * 2048 + dir * 1024 + col] = hn;
      }
    }
    __syncthreads();
    cur ^= 1;
  }

  // Final states: state[dir] = concat(h1_final, h2_final) along features.
  float* st = a.state[dir];
  const unsigned short* h1f = lds + cur * HSZ;
  const unsigned short* h2f = lds + (2 + cur) * HSZ;
  for (int i = tid; i < HSZ; i += 1024) {
    const int row = i >> 10, c = i & 1023;
    st[row * 2048 + c]        = bf2f(h1f[i]);
    st[row * 2048 + 1024 + c] = bf2f(h2f[i]);
  }
}

extern "C" void kernel_launch(void* const* d_in, const int* in_sizes, int n_in,
                              void* d_out, int out_size, void* d_ws, size_t ws_size,
                              hipStream_t stream) {
  const int*   tokens = (const int*)d_in[0];
  const float* W1f = (const float*)d_in[1];
  const float* R1f = (const float*)d_in[2];
  const float* b1f = (const float*)d_in[3];
  const float* W2f = (const float*)d_in[4];
  const float* R2f = (const float*)d_in[5];
  const float* b2f = (const float*)d_in[6];
  const float* W1b = (const float*)d_in[7];
  const float* R1b = (const float*)d_in[8];
  const float* b1b = (const float*)d_in[9];
  const float* W2b = (const float*)d_in[10];
  const float* R2b = (const float*)d_in[11];
  const float* b2b = (const float*)d_in[12];

  // Workspace: 6 packed bf16 weight images (1024x3072 each) = 36 MB.
  unsigned short* ws = (unsigned short*)d_ws;
  const size_t msz = (size_t)UU * G3;
  unsigned short* pR1f = ws + 0 * msz;
  unsigned short* pW2f = ws + 1 * msz;
  unsigned short* pR2f = ws + 2 * msz;
  unsigned short* pR1b = ws + 3 * msz;
  unsigned short* pW2b = ws + 4 * msz;
  unsigned short* pR2b = ws + 5 * msz;

  const int packBlocks = KTILES * NTILES;   // 6144
  pack_weights<<<packBlocks, 32, 0, stream>>>(R1f, pR1f);
  pack_weights<<<packBlocks, 32, 0, stream>>>(W2f, pW2f);
  pack_weights<<<packBlocks, 32, 0, stream>>>(R2f, pR2f);
  pack_weights<<<packBlocks, 32, 0, stream>>>(R1b, pR1b);
  pack_weights<<<packBlocks, 32, 0, stream>>>(W2b, pW2b);
  pack_weights<<<packBlocks, 32, 0, stream>>>(R2b, pR2b);

  Args a{};
  a.tokens = tokens;
  a.W1[0] = W1f;  a.W1[1] = W1b;
  a.b1[0] = b1f;  a.b1[1] = b1b;
  a.b2[0] = b2f;  a.b2[1] = b2b;
  a.PR1[0] = pR1f; a.PR1[1] = pR1b;
  a.PW2[0] = pW2f; a.PW2[1] = pW2b;
  a.PR2[0] = pR2f; a.PR2[1] = pR2b;
  a.out = (float*)d_out;
  a.state[0] = (float*)d_out + (size_t)BB * SS * 2048;          // state1 (fwd)
  a.state[1] = a.state[0] + (size_t)BB * 2048;                  // state2 (bwd)

  const size_t ldsBytes = 4 * (size_t)HSZ * sizeof(unsigned short); // 256 KB
  (void)hipFuncSetAttribute((const void*)gru_encoder,
                            hipFuncAttributeMaxDynamicSharedMemorySize,
                            (int)ldsBytes);
  gru_encoder<<<2, 1024, ldsBytes, stream>>>(a);
}
